// AdvancedCardiomyocyteGNN_84945863180762
// MI455X (gfx1250) — compile-verified, matrix operand order checked by reading, not run on hardware
//
#include <hip/hip_runtime.h>
#include <hip/hip_bf16.h>
#include <math.h>

// ---------------- problem constants ----------------
#define NN     10000          // nodes
#define NE     160000         // edges
#define ELN    (NE + NN)      // edges + self loops
#define FIN    2000
#define KPX    2016           // FIN padded to mult of 32
#define HIDC   128

typedef __attribute__((ext_vector_type(16))) __bf16 v16bf;
typedef __attribute__((ext_vector_type(8)))  __bf16 v8bf;
typedef __attribute__((ext_vector_type(8)))  float  v8f;

// ================= pack kernels =================

// fp32 [M,K] -> bf16 [M,Kp] row major, zero-pad K..Kp
__global__ void pack_a_bf16(const float* __restrict__ src, __bf16* __restrict__ dst,
                            int M, int K, int Kp) {
  long long i = (long long)blockIdx.x * blockDim.x + threadIdx.x;
  long long total = (long long)M * Kp;
  if (i >= total) return;
  int kp = (int)(i % Kp);
  long long n = i / Kp;
  float v = (kp < K) ? src[n * K + kp] : 0.0f;
  dst[i] = (__bf16)v;
}

// fp32 W [K,Nreal] row-major -> bf16 packed B fragments:
// Bp[((kt*Npad + n)*32 + j)], j order per 32-K block: [0..7,16..23,8..15,24..31]
__global__ void pack_b_bf16(const float* __restrict__ W, __bf16* __restrict__ Bp,
                            int K, int Kp, int Nreal, int Npad) {
  long long i = (long long)blockIdx.x * blockDim.x + threadIdx.x;
  long long total = (long long)Kp * Npad;
  if (i >= total) return;
  int j = (int)(i & 31);
  long long t = i >> 5;
  int n  = (int)(t % Npad);
  int kt = (int)(t / Npad);
  int grp = j >> 3;
  int k = kt * 32 + j + (grp == 1 ? 8 : (grp == 2 ? -8 : 0));
  float v = (k < K && n < Nreal) ? W[(long long)k * Nreal + n] : 0.0f;
  Bp[i] = (__bf16)v;
}

// concat(x1_gcn[M,128], x2_gcn[M,64]) -> bf16 [M,192]
__global__ void pack_concat_bf16(const float* __restrict__ a, const float* __restrict__ b,
                                 __bf16* __restrict__ dst, int M) {
  long long i = (long long)blockIdx.x * blockDim.x + threadIdx.x;
  long long total = (long long)M * 192;
  if (i >= total) return;
  int c = (int)(i % 192);
  long long n = i / 192;
  float v = (c < 128) ? a[n * 128 + c] : b[n * 64 + (c - 128)];
  dst[i] = (__bf16)v;
}

// ================= WMMA GEMM =================
// C[M,Nreal] = A[M,Kp](bf16,row-major) x B(packed bf16).
// Block = 256 threads = 8 waves; block tile 256x64.
// Each wave: 32 rows x 64 cols = 2x4 register blocking -> 8 WMMA per k-step.
// B tile (4KB) double-buffered in LDS via GLOBAL_LOAD_ASYNC_TO_LDS_B128 (ASYNCcnt).
__global__ __launch_bounds__(256) void gemm_bf16_wmma(
    const __bf16* __restrict__ A, const __bf16* __restrict__ Bp,
    float* __restrict__ C, const float* __restrict__ bias,
    int M, int Kp, int Npad, int Nreal, int relu) {
  __shared__ __align__(32) __bf16 Bs[2][64 * 32];   // 2 x 4 KB

  const int lane = threadIdx.x & 31;
  const int wave = threadIdx.x >> 5;
  const int half = lane >> 4;
  const int l16  = lane & 15;
  const int rowBase = blockIdx.x * 256 + wave * 32;
  const int colBase = blockIdx.y * 64;

  v8f acc[2][4];
#pragma unroll
  for (int r = 0; r < 2; ++r)
#pragma unroll
    for (int n = 0; n < 4; ++n) acc[r][n] = (v8f){};

  int r0 = rowBase + l16;      if (r0 >= M) r0 = M - 1;   // clamp, stores masked
  int r1 = rowBase + 16 + l16; if (r1 >= M) r1 = M - 1;
  const __bf16* Arow0 = A + (long long)r0 * Kp + half * 8;
  const __bf16* Arow1 = A + (long long)r1 * Kp + half * 8;

  const int ksteps = Kp >> 5;

  // async-stage one 4KB B tile (64 cols x 32 k) into LDS buffer `buf`.
  // IMPORTANT: the LDS address operand is derived from &Bs (ptrtoint) so the
  // shared array escapes -> the "memory" clobber forces the compiler to treat
  // Bs as written (otherwise the B-fragment ds_loads get folded to undef).
  auto stage = [&](int kt, int buf) {
    const uint4* g = (const uint4*)(Bp + ((long long)kt * Npad + colBase) * 32) + threadIdx.x;
    unsigned lds = (unsigned)(uintptr_t)(&Bs[buf][threadIdx.x * 8]);  // low 32 bits = LDS offset
    unsigned long long ga = (unsigned long long)(uintptr_t)g;
    asm volatile("global_load_async_to_lds_b128 %0, %1, off"
                 :: "v"(lds), "v"(ga) : "memory");
  };

  stage(0, 0);
  for (int kt = 0; kt < ksteps; ++kt) {
    asm volatile("s_wait_asynccnt 0x0" ::: "memory");  // my tile writes done
    __syncthreads();                                    // everyone's writes/reads done
    if (kt + 1 < ksteps) stage(kt + 1, (kt + 1) & 1);

    // A fragments: two contiguous 16B chunks each
    union { v16bf v; v8bf h[2]; } a0, a1;
    const __bf16* ak0 = Arow0 + kt * 32;
    const __bf16* ak1 = Arow1 + kt * 32;
    a0.h[0] = *(const v8bf*)(ak0);
    a0.h[1] = *(const v8bf*)(ak0 + 16);
    a1.h[0] = *(const v8bf*)(ak1);
    a1.h[1] = *(const v8bf*)(ak1 + 16);

    const v16bf* bsv = (const v16bf*)(Bs[kt & 1]);
#pragma unroll
    for (int nt = 0; nt < 4; ++nt) {
      v16bf bf = bsv[(nt * 16 + l16) * 2 + half];
      acc[0][nt] = __builtin_amdgcn_wmma_f32_16x16x32_bf16(
          false, a0.v, false, bf, (short)0, acc[0][nt], false, false);
      acc[1][nt] = __builtin_amdgcn_wmma_f32_16x16x32_bf16(
          false, a1.v, false, bf, (short)0, acc[1][nt], false, false);
    }
  }

  // epilogue: optional bias + relu, masked store
#pragma unroll
  for (int rt = 0; rt < 2; ++rt) {
#pragma unroll
    for (int nt = 0; nt < 4; ++nt) {
      int col = colBase + nt * 16 + l16;
      float badd = 0.0f;
      if (bias != nullptr && col < Nreal) badd = bias[col];
#pragma unroll
      for (int r = 0; r < 8; ++r) {
        int row = rowBase + rt * 16 + r + half * 8;
        if (row < M && col < Nreal) {
          float v = acc[rt][nt][r] + badd;
          if (relu) v = fmaxf(v, 0.0f);
          C[(long long)row * Nreal + col] = v;
        }
      }
    }
  }
}

// ================= graph / elementwise kernels =================

__global__ void zero_f32(float* __restrict__ p, long long n) {
  long long i = (long long)blockIdx.x * blockDim.x + threadIdx.x;
  if (i < n) p[i] = 0.0f;
}

__global__ void recip_f32(float* __restrict__ p, long long n) {
  long long i = (long long)blockIdx.x * blockDim.x + threadIdx.x;
  if (i < n) p[i] = 1.0f / p[i];
}

// per (node, head): alpha_src/dst dot products over 128 channels
__global__ void node_alpha(const float* __restrict__ Hf, const float* __restrict__ a_src,
                           const float* __restrict__ a_dst, float* __restrict__ as,
                           float* __restrict__ ad, int H) {
  int i = blockIdx.x * blockDim.x + threadIdx.x;
  if (i >= NN * H) return;
  int h = i % H;
  int n = i / H;
  const float4* hr = (const float4*)(Hf + (long long)n * H * HIDC + h * HIDC);
  const float4* s4 = (const float4*)(a_src + h * HIDC);
  const float4* d4 = (const float4*)(a_dst + h * HIDC);
  float ss = 0.0f, dd = 0.0f;
#pragma unroll 4
  for (int c = 0; c < 32; ++c) {
    float4 v = hr[c], s = s4[c], d = d4[c];
    ss += v.x * s.x + v.y * s.y + v.z * s.z + v.w * s.w;
    dd += v.x * d.x + v.y * d.y + v.z * d.z + v.w * d.w;
  }
  as[i] = ss;
  ad[i] = dd;
}

// pass 1: exp(leaky_relu(alpha)) per (edge, head) + segment-sum of denominators.
// (e/Σe is invariant to the max-shift, so segment_max is skipped.)
__global__ void edge_softmax_p1(const int* __restrict__ ei, const float* __restrict__ as,
                                const float* __restrict__ ad, float* __restrict__ ew,
                                float* __restrict__ den, int H) {
  int e = blockIdx.x * blockDim.x + threadIdx.x;
  if (e >= ELN) return;
  int src = (e < NE) ? ei[e] : (e - NE);
  int dst = (e < NE) ? ei[NE + e] : (e - NE);
  for (int h = 0; h < H; ++h) {
    float a = as[src * H + h] + ad[dst * H + h];
    a = (a > 0.0f) ? a : 0.2f * a;
    float ex = __expf(a);
    ew[(long long)e * H + h] = ex;
    atomicAdd(&den[dst * H + h], ex);
  }
}

// pass 2: head-mean weighted aggregation, thread per (edge, channel).
// rden holds reciprocal denominators.
__global__ void gat_agg(const int* __restrict__ ei, const float* __restrict__ Hf,
                        const float* __restrict__ ew, const float* __restrict__ rden,
                        float* __restrict__ out, int H, float invH) {
  long long t = (long long)blockIdx.x * blockDim.x + threadIdx.x;
  if (t >= (long long)ELN * HIDC) return;
  int c = (int)(t & (HIDC - 1));
  int e = (int)(t >> 7);
  int src = (e < NE) ? ei[e] : (e - NE);
  int dst = (e < NE) ? ei[NE + e] : (e - NE);
  float acc = 0.0f;
  for (int h = 0; h < H; ++h) {
    float w = ew[(long long)e * H + h] * rden[dst * H + h];
    acc += w * Hf[(long long)src * H * HIDC + h * HIDC + c];
  }
  atomicAdd(&out[(long long)dst * HIDC + c], acc * invH);
}

// degree (with self loops) and D^-1/2
__global__ void deg_kernel(const int* __restrict__ ei, float* __restrict__ deg) {
  int e = blockIdx.x * blockDim.x + threadIdx.x;
  if (e >= ELN) return;
  int dst = (e < NE) ? ei[NE + e] : (e - NE);
  atomicAdd(&deg[dst], 1.0f);
}
__global__ void dinv_kernel(const float* __restrict__ deg, float* __restrict__ dinv) {
  int i = blockIdx.x * blockDim.x + threadIdx.x;
  if (i >= NN) return;
  float d = deg[i];
  dinv[i] = (d > 0.0f) ? rsqrtf(d) : 0.0f;
}

// GCN normalized message aggregation, thread per (edge, channel); C power of 2
__global__ void gcn_agg(const int* __restrict__ ei, const float* __restrict__ pre,
                        const float* __restrict__ dinv, float* __restrict__ out,
                        int C, int logC) {
  long long t = (long long)blockIdx.x * blockDim.x + threadIdx.x;
  if (t >= (long long)ELN * C) return;
  int c = (int)(t & (C - 1));
  int e = (int)(t >> logC);
  int src = (e < NE) ? ei[e] : (e - NE);
  int dst = (e < NE) ? ei[NE + e] : (e - NE);
  float m = pre[(long long)src * C + c] * dinv[src] * dinv[dst];
  atomicAdd(&out[(long long)dst * C + c], m);
}

__global__ void bias_relu(float* __restrict__ x, const float* __restrict__ b,
                          long long n, int Cmask) {
  long long i = (long long)blockIdx.x * blockDim.x + threadIdx.x;
  if (i >= n) return;
  int c = (int)(i & Cmask);
  x[i] = fmaxf(x[i] + b[c], 0.0f);
}

// wave-per-node LayerNorm: out = [relu?](LN(A + Badd? + bias?) * g + be)
template <int C>
__global__ __launch_bounds__(256) void ln_kernel(
    const float* __restrict__ A, const float* __restrict__ Badd,
    const float* __restrict__ bias, const float* __restrict__ g,
    const float* __restrict__ be, float* __restrict__ out, int relu) {
  const int lane = threadIdx.x & 31;
  const int wave = threadIdx.x >> 5;
  int n = blockIdx.x * 8 + wave;
  if (n >= NN) return;
  constexpr int PER = C / 32;
  float v[PER];
  float s = 0.0f, sq = 0.0f;
#pragma unroll
  for (int i = 0; i < PER; ++i) {
    int c = lane + i * 32;
    float x = A[(long long)n * C + c];
    if (Badd) x += Badd[(long long)n * C + c];
    if (bias) x += bias[c];
    v[i] = x; s += x; sq += x * x;
  }
#pragma unroll
  for (int m = 16; m >= 1; m >>= 1) {
    s  += __shfl_xor(s,  m, 32);
    sq += __shfl_xor(sq, m, 32);
  }
  float mu  = s / (float)C;
  float var = sq / (float)C - mu * mu;
  float rs  = rsqrtf(var + 1e-5f);
#pragma unroll
  for (int i = 0; i < PER; ++i) {
    int c = lane + i * 32;
    float y = (v[i] - mu) * rs * g[c] + be[c];
    if (relu) y = fmaxf(y, 0.0f);
    out[(long long)n * C + c] = y;
  }
}

// ================= host orchestration =================

static inline long long cdivll(long long a, long long b) { return (a + b - 1) / b; }

extern "C" void kernel_launch(void* const* d_in, const int* in_sizes, int n_in,
                              void* d_out, int out_size, void* d_ws, size_t ws_size,
                              hipStream_t stream) {
  (void)in_sizes; (void)n_in; (void)out_size; (void)ws_size;
  const float* x      = (const float*)d_in[0];
  const int*   ei     = (const int*)d_in[1];
  const float* W_gat1 = (const float*)d_in[2];
  const float* a_s1   = (const float*)d_in[3];
  const float* a_d1   = (const float*)d_in[4];
  const float* b_gat1 = (const float*)d_in[5];
  const float* W_gcn1 = (const float*)d_in[6];
  const float* b_gcn1 = (const float*)d_in[7];
  const float* W_gat2 = (const float*)d_in[8];
  const float* a_s2   = (const float*)d_in[9];
  const float* a_d2   = (const float*)d_in[10];
  const float* b_gat2 = (const float*)d_in[11];
  const float* W_gcn2 = (const float*)d_in[12];
  const float* b_gcn2 = (const float*)d_in[13];
  const float* W_skip = (const float*)d_in[14];
  const float* b_skip = (const float*)d_in[15];
  const float* W_fuse = (const float*)d_in[16];
  const float* b_fuse = (const float*)d_in[17];
  const float* W_c1   = (const float*)d_in[18];
  const float* b_c1   = (const float*)d_in[19];
  const float* W_c2   = (const float*)d_in[20];
  const float* b_c2   = (const float*)d_in[21];
  const float* W_c3   = (const float*)d_in[22];
  const float* b_c3   = (const float*)d_in[23];
  const float* g1 = (const float*)d_in[24];
  const float* be1 = (const float*)d_in[25];
  const float* g2 = (const float*)d_in[26];
  const float* be2 = (const float*)d_in[27];
  const float* g3 = (const float*)d_in[28];
  const float* be3 = (const float*)d_in[29];
  float* out = (float*)d_out;

  // ---- bump allocator over d_ws ----
  char* base = (char*)d_ws;
  size_t off = 0;
  auto alloc = [&](size_t bytes) -> void* {
    void* p = base + off;
    off = (off + bytes + 255) & ~(size_t)255;
    return p;
  };
  __bf16* xb    = (__bf16*)alloc((size_t)NN * KPX * 2);        // 40.3 MB (arena A)
  __bf16* Wg1p  = (__bf16*)alloc((size_t)KPX * 768 * 2);
  __bf16* Wskp  = (__bf16*)alloc((size_t)KPX * 64 * 2);
  float*  H1    = (float*)alloc((size_t)NN * 768 * 4);         // 30.7 MB
  float*  skp   = (float*)alloc((size_t)NN * 64 * 4);
  float*  as1   = (float*)alloc((size_t)NN * 6 * 4);
  float*  ad1   = (float*)alloc((size_t)NN * 6 * 4);
  float*  ew1   = (float*)alloc((size_t)ELN * 6 * 4);
  float*  den1  = (float*)alloc((size_t)NN * 6 * 4);
  float*  gat1o = (float*)alloc((size_t)NN * HIDC * 4);
  float*  x1gat = (float*)alloc((size_t)NN * HIDC * 4);
  __bf16* x1gatb= (__bf16*)alloc((size_t)NN * HIDC * 2);
  __bf16* Wgc1p = (__bf16*)alloc((size_t)HIDC * HIDC * 2);
  float*  gcn1p = (float*)alloc((size_t)NN * HIDC * 4);
  float*  deg   = (float*)alloc((size_t)NN * 4);
  float*  dinv  = (float*)alloc((size_t)NN * 4);
  float*  x1gcn = (float*)alloc((size_t)NN * HIDC * 4);
  __bf16* x1gcnb= (__bf16*)alloc((size_t)NN * HIDC * 2);
  __bf16* Wg2p  = (__bf16*)alloc((size_t)HIDC * 512 * 2);
  float*  as2   = (float*)alloc((size_t)NN * 4 * 4);
  float*  ad2   = (float*)alloc((size_t)NN * 4 * 4);
  float*  ew2   = (float*)alloc((size_t)ELN * 4 * 4);
  float*  den2  = (float*)alloc((size_t)NN * 4 * 4);
  float*  gat2o = (float*)alloc((size_t)NN * HIDC * 4);
  float*  x2gat = (float*)alloc((size_t)NN * HIDC * 4);
  __bf16* x2gatb= (__bf16*)alloc((size_t)NN * HIDC * 2);
  __bf16* Wgc2p = (__bf16*)alloc((size_t)HIDC * 64 * 2);
  float*  gcn2p = (float*)alloc((size_t)NN * 64 * 4);
  float*  x2gcn = (float*)alloc((size_t)NN * 64 * 4);
  __bf16* catb  = (__bf16*)alloc((size_t)NN * 192 * 2);
  __bf16* Wfup  = (__bf16*)alloc((size_t)192 * 64 * 2);
  float*  fuseo = (float*)alloc((size_t)NN * 64 * 4);
  float*  finl  = (float*)alloc((size_t)NN * 64 * 4);
  __bf16* finlb = (__bf16*)alloc((size_t)NN * 64 * 2);
  __bf16* Wc1p  = (__bf16*)alloc((size_t)64 * 64 * 2);
  float*  h1o   = (float*)alloc((size_t)NN * 32 * 4);
  __bf16* h1b   = (__bf16*)alloc((size_t)NN * 32 * 2);
  __bf16* Wc2p  = (__bf16*)alloc((size_t)32 * 64 * 2);
  float*  h2o   = (float*)alloc((size_t)NN * 16 * 4);
  __bf16* h2b   = (__bf16*)alloc((size_t)NN * 32 * 2);
  __bf16* Wc3p  = (__bf16*)alloc((size_t)32 * 64 * 2);
  // H2 aliases the x-bf16 arena (xb is dead once the two big GEMMs are done)
  float*  H2    = (float*)xb;                                   // 20.5 MB <= 40.3 MB

  const int TPB = 256;
  auto g1d = [&](long long n) { return (unsigned)cdivll(n, TPB); };
  auto gemmGrid = [&](int Npad) { return dim3((NN + 255) / 256, Npad / 64); };

  // ---- stage 1: big GEMMs off x ----
  pack_a_bf16<<<g1d((long long)NN * KPX), TPB, 0, stream>>>(x, xb, NN, FIN, KPX);
  pack_b_bf16<<<g1d((long long)KPX * 768), TPB, 0, stream>>>(W_gat1, Wg1p, FIN, KPX, 768, 768);
  pack_b_bf16<<<g1d((long long)KPX * 64), TPB, 0, stream>>>(W_skip, Wskp, FIN, KPX, 64, 64);
  gemm_bf16_wmma<<<gemmGrid(768), TPB, 0, stream>>>(xb, Wg1p, H1, nullptr, NN, KPX, 768, 768, 0);
  gemm_bf16_wmma<<<gemmGrid(64), TPB, 0, stream>>>(xb, Wskp, skp, b_skip, NN, KPX, 64, 64, 1);

  // ---- GAT conv 1 (H=6) ----
  node_alpha<<<g1d((long long)NN * 6), TPB, 0, stream>>>(H1, a_s1, a_d1, as1, ad1, 6);
  zero_f32<<<g1d((long long)NN * 6), TPB, 0, stream>>>(den1, (long long)NN * 6);
  edge_softmax_p1<<<g1d(ELN), TPB, 0, stream>>>(ei, as1, ad1, ew1, den1, 6);
  recip_f32<<<g1d((long long)NN * 6), TPB, 0, stream>>>(den1, (long long)NN * 6);
  zero_f32<<<g1d((long long)NN * HIDC), TPB, 0, stream>>>(gat1o, (long long)NN * HIDC);
  gat_agg<<<g1d((long long)ELN * HIDC), TPB, 0, stream>>>(ei, H1, ew1, den1, gat1o, 6, 1.0f / 6.0f);
  ln_kernel<128><<<(NN + 7) / 8, TPB, 0, stream>>>(gat1o, nullptr, b_gat1, g1, be1, x1gat, 1);

  // ---- GCN conv 1 ----
  pack_a_bf16<<<g1d((long long)NN * HIDC), TPB, 0, stream>>>(x1gat, x1gatb, NN, HIDC, HIDC);
  pack_b_bf16<<<g1d((long long)HIDC * HIDC), TPB, 0, stream>>>(W_gcn1, Wgc1p, HIDC, HIDC, HIDC, HIDC);
  gemm_bf16_wmma<<<gemmGrid(HIDC), TPB, 0, stream>>>(x1gatb, Wgc1p, gcn1p, nullptr, NN, HIDC, HIDC, HIDC, 0);
  zero_f32<<<g1d(NN), TPB, 0, stream>>>(deg, NN);
  deg_kernel<<<g1d(ELN), TPB, 0, stream>>>(ei, deg);
  dinv_kernel<<<g1d(NN), TPB, 0, stream>>>(deg, dinv);
  zero_f32<<<g1d((long long)NN * HIDC), TPB, 0, stream>>>(x1gcn, (long long)NN * HIDC);
  gcn_agg<<<g1d((long long)ELN * HIDC), TPB, 0, stream>>>(ei, gcn1p, dinv, x1gcn, HIDC, 7);
  bias_relu<<<g1d((long long)NN * HIDC), TPB, 0, stream>>>(x1gcn, b_gcn1, (long long)NN * HIDC, HIDC - 1);

  // ---- GAT conv 2 (H=4) ----
  pack_a_bf16<<<g1d((long long)NN * HIDC), TPB, 0, stream>>>(x1gcn, x1gcnb, NN, HIDC, HIDC);
  pack_b_bf16<<<g1d((long long)HIDC * 512), TPB, 0, stream>>>(W_gat2, Wg2p, HIDC, HIDC, 512, 512);
  gemm_bf16_wmma<<<gemmGrid(512), TPB, 0, stream>>>(x1gcnb, Wg2p, H2, nullptr, NN, HIDC, 512, 512, 0);
  node_alpha<<<g1d((long long)NN * 4), TPB, 0, stream>>>(H2, a_s2, a_d2, as2, ad2, 4);
  zero_f32<<<g1d((long long)NN * 4), TPB, 0, stream>>>(den2, (long long)NN * 4);
  edge_softmax_p1<<<g1d(ELN), TPB, 0, stream>>>(ei, as2, ad2, ew2, den2, 4);
  recip_f32<<<g1d((long long)NN * 4), TPB, 0, stream>>>(den2, (long long)NN * 4);
  zero_f32<<<g1d((long long)NN * HIDC), TPB, 0, stream>>>(gat2o, (long long)NN * HIDC);
  gat_agg<<<g1d((long long)ELN * HIDC), TPB, 0, stream>>>(ei, H2, ew2, den2, gat2o, 4, 0.25f);
  ln_kernel<128><<<(NN + 7) / 8, TPB, 0, stream>>>(gat2o, nullptr, b_gat2, g2, be2, x2gat, 1);

  // ---- GCN conv 2 (out 64) ----
  pack_a_bf16<<<g1d((long long)NN * HIDC), TPB, 0, stream>>>(x2gat, x2gatb, NN, HIDC, HIDC);
  pack_b_bf16<<<g1d((long long)HIDC * 64), TPB, 0, stream>>>(W_gcn2, Wgc2p, HIDC, HIDC, 64, 64);
  gemm_bf16_wmma<<<gemmGrid(64), TPB, 0, stream>>>(x2gatb, Wgc2p, gcn2p, nullptr, NN, HIDC, 64, 64, 0);
  zero_f32<<<g1d((long long)NN * 64), TPB, 0, stream>>>(x2gcn, (long long)NN * 64);
  gcn_agg<<<g1d((long long)ELN * 64), TPB, 0, stream>>>(ei, gcn2p, dinv, x2gcn, 64, 6);
  bias_relu<<<g1d((long long)NN * 64), TPB, 0, stream>>>(x2gcn, b_gcn2, (long long)NN * 64, 63);

  // ---- fuse + final LN ----
  pack_concat_bf16<<<g1d((long long)NN * 192), TPB, 0, stream>>>(x1gcn, x2gcn, catb, NN);
  pack_b_bf16<<<g1d((long long)192 * 64), TPB, 0, stream>>>(W_fuse, Wfup, 192, 192, 64, 64);
  gemm_bf16_wmma<<<gemmGrid(64), TPB, 0, stream>>>(catb, Wfup, fuseo, b_fuse, NN, 192, 64, 64, 1);
  ln_kernel<64><<<(NN + 7) / 8, TPB, 0, stream>>>(fuseo, skp, nullptr, g3, be3, finl, 0);

  // ---- classifier MLP (all WMMA) ----
  pack_a_bf16<<<g1d((long long)NN * 64), TPB, 0, stream>>>(finl, finlb, NN, 64, 64);
  pack_b_bf16<<<g1d((long long)64 * 64), TPB, 0, stream>>>(W_c1, Wc1p, 64, 64, 32, 64);
  gemm_bf16_wmma<<<gemmGrid(64), TPB, 0, stream>>>(finlb, Wc1p, h1o, b_c1, NN, 64, 64, 32, 1);
  pack_a_bf16<<<g1d((long long)NN * 32), TPB, 0, stream>>>(h1o, h1b, NN, 32, 32);
  pack_b_bf16<<<g1d((long long)32 * 64), TPB, 0, stream>>>(W_c2, Wc2p, 32, 32, 16, 64);
  gemm_bf16_wmma<<<gemmGrid(64), TPB, 0, stream>>>(h1b, Wc2p, h2o, b_c2, NN, 32, 64, 16, 1);
  pack_a_bf16<<<g1d((long long)NN * 32), TPB, 0, stream>>>(h2o, h2b, NN, 16, 32);
  pack_b_bf16<<<g1d((long long)32 * 64), TPB, 0, stream>>>(W_c3, Wc3p, 16, 32, 5, 64);
  gemm_bf16_wmma<<<gemmGrid(64), TPB, 0, stream>>>(h2b, Wc3p, out, b_c3, NN, 32, 64, 5, 0);
}
